// seg_atn_47845935677672
// MI455X (gfx1250) — matrix operand order repaired and checked by reading, not compiled, not run on hardware
//
#include <hip/hip_runtime.h>
#include <math.h>

#define DK 512
#define OUTD 512

typedef __attribute__((ext_vector_type(16))) _Float16 v16h;
typedef __attribute__((ext_vector_type(8)))  float    v8f;

__device__ __forceinline__ int lower_bound(const int* __restrict__ td, int R, int v) {
    int lo = 0, hi = R;
    while (lo < hi) {
        int mid = (lo + hi) >> 1;
        if (td[mid] < v) lo = mid + 1; else hi = mid;
    }
    return lo;
}

// ---------------------------------------------------------------------------
// Kernel 1: fused score + online segment softmax + weighted aggregation.
// One 256-thread block per segment (query). Each thread owns 2 of the 512
// dims. K is streamed from HBM exactly once (512 MB total across the grid).
// Online softmax: running max m, running denom d, rescaled accumulators.
// ---------------------------------------------------------------------------
__global__ __launch_bounds__(256) void seg_attn_fused(
    const float* __restrict__ Q, const float* __restrict__ Kmat,
    const int* __restrict__ td, float* __restrict__ ctx, int R)
{
    const int seg  = blockIdx.x;
    const int t    = threadIdx.x;        // 0..255
    const int lane = t & 31;             // wave32
    const int wid  = t >> 5;             // 8 waves / block

    __shared__ float red[8];

    const int s = lower_bound(td, R, seg);
    const int e = lower_bound(td, R, seg + 1);

    const float q0 = Q[(size_t)seg * DK + t];
    const float q1 = Q[(size_t)seg * DK + t + 256];

    float a0 = 0.f, a1 = 0.f;
    float m = -INFINITY, d = 0.f;

    for (int i = s; i < e; ++i) {
        const float* kr = Kmat + (size_t)i * DK;
        const float k0 = kr[t];
        const float k1 = kr[t + 256];

        // block-wide dot product <Q[seg], K[i]>
        float p = fmaf(q0, k0, q1 * k1);
        #pragma unroll
        for (int off = 16; off; off >>= 1) p += __shfl_xor(p, off, 32);
        if (lane == 0) red[wid] = p;
        __syncthreads();
        float sc = red[0] + red[1] + red[2] + red[3]
                 + red[4] + red[5] + red[6] + red[7];
        __syncthreads();   // red[] reused next iteration

        // online softmax update
        const float mnew  = fmaxf(m, sc);
        const float scale = __expf(m - mnew);   // first iter: exp(-inf) = 0
        const float w     = __expf(sc - mnew);
        a0 = fmaf(a0, scale, w * k0);
        a1 = fmaf(a1, scale, w * k1);
        d  = fmaf(d,  scale, w);
        m  = mnew;
    }

    const float rd = (d > 0.f) ? (1.f / d) : 0.f;   // empty segment -> ctx = 0
    ctx[(size_t)seg * DK + t]       = a0 * rd;
    ctx[(size_t)seg * DK + t + 256] = a1 * rd;
}

// ---------------------------------------------------------------------------
// Kernel 2: out = ctx @ W_o + b_o via v_wmma_f32_16x16x32_f16.
// Each wave computes a 64x16 output strip: 4 M-tiles sharing one B fragment
// per k-step (4 WMMAs per 16 strided B loads; 4 independent accumulation
// chains hide the WMMA->WMMA hazard window). All 8 waves in a block share the
// same N-tile of W_o (1 MB matrix, L2/L0 resident).
//   grid = (OUT/16, Mtiles/32), block = 256 (8 waves, 4 M-tiles each)
// ---------------------------------------------------------------------------
__global__ __launch_bounds__(256) void out_gemm_wmma(
    const float* __restrict__ ctx, const float* __restrict__ W_o,
    const float* __restrict__ b_o, float* __restrict__ out, int Mtiles)
{
    const int tileN  = blockIdx.x;                              // 0..OUT/16-1
    const int tileM0 = (blockIdx.y * 8 + (threadIdx.x >> 5)) * 4;
    if (tileM0 >= Mtiles) return;                               // wave-uniform

    const int lane = threadIdx.x & 31;
    const int h    = lane >> 4;      // half-wave select
    const int mn   = lane & 15;      // row (A) / col (B,D)

    // A row pointers for the 4 M-tiles this wave owns
    const float* arow0 = ctx + (size_t)(tileM0 * 16 + mn) * DK;
    const float* arow1 = arow0 + (size_t)16 * DK;
    const float* arow2 = arow0 + (size_t)32 * DK;
    const float* arow3 = arow0 + (size_t)48 * DK;
    const int    ng    = tileN * 16 + mn;                       // output column

    v8f c0 = {}, c1 = {}, c2 = {}, c3 = {};

    for (int k0 = 0; k0 < DK; k0 += 32) {
        // B tile 32x16 f16: b[j] = B[k0 + h*8 + j][n], b[8+j] = B[k0+16+h*8+j][n]   (shared by 4 WMMAs)
        v16h b;
        const float* bp = W_o + (size_t)(k0 + h * 8) * OUTD + ng;
        #pragma unroll
        for (int j = 0; j < 16; ++j) {
            b[j] = (_Float16)bp[(size_t)(j + (j & 8)) * OUTD];
        }

        // A tiles 16x32 f16: a[j] = A[m][k0 + h*8 + j], a[8+j] = A[m][k0+16+h*8+j]
        v16h a0, a1, a2, a3;
        #pragma unroll
        for (int j = 0; j < 8; ++j) {
            const int klo = k0 + h * 8 + j;
            const int khi = k0 + 16 + h * 8 + j;
            a0[j] = (_Float16)arow0[klo];  a0[8 + j] = (_Float16)arow0[khi];
            a1[j] = (_Float16)arow1[klo];  a1[8 + j] = (_Float16)arow1[khi];
            a2[j] = (_Float16)arow2[klo];  a2[8 + j] = (_Float16)arow2[khi];
            a3[j] = (_Float16)arow3[klo];  a3[8 + j] = (_Float16)arow3[khi];
        }

        c0 = __builtin_amdgcn_wmma_f32_16x16x32_f16(false, a0, false, b, (short)0, c0, false, false);
        c1 = __builtin_amdgcn_wmma_f32_16x16x32_f16(false, a1, false, b, (short)0, c1, false, false);
        c2 = __builtin_amdgcn_wmma_f32_16x16x32_f16(false, a2, false, b, (short)0, c2, false, false);
        c3 = __builtin_amdgcn_wmma_f32_16x16x32_f16(false, a3, false, b, (short)0, c3, false, false);
    }

    // D layout: c[r] = D[r + 8*h][mn]; add bias and store f32
    const float bias = b_o[ng];
    #pragma unroll
    for (int r = 0; r < 8; ++r) {
        const size_t mg = (size_t)(tileM0 * 16 + r + 8 * h);
        out[(mg +  0) * OUTD + ng] = c0[r] + bias;
        out[(mg + 16) * OUTD + ng] = c1[r] + bias;
        out[(mg + 32) * OUTD + ng] = c2[r] + bias;
        out[(mg + 48) * OUTD + ng] = c3[r] + bias;
    }
}

extern "C" void kernel_launch(void* const* d_in, const int* in_sizes, int n_in,
                              void* d_out, int out_size, void* d_ws, size_t ws_size,
                              hipStream_t stream) {
    const float* Q   = (const float*)d_in[0];
    const float* K   = (const float*)d_in[1];
    const int*   td  = (const int*)d_in[2];
    const float* W_o = (const float*)d_in[3];
    const float* b_o = (const float*)d_in[4];
    float*       out = (float*)d_out;

    const int B = in_sizes[0] / DK;   // 16384
    const int R = in_sizes[2];        // 262144

    float* ctx = (float*)d_ws;        // B*DK f32 = 32 MB, fully rewritten each call

    // Pass 1: single-pass fused segment attention (K read once from HBM)
    seg_attn_fused<<<B, 256, 0, stream>>>(Q, K, td, ctx, R);

    // Pass 2: WMMA output projection (wave = 64x16 strip, 4 WMMAs per B fragment)
    const int Mtiles = B / 16;                        // 1024
    dim3 grid(OUTD / 16, (Mtiles + 31) / 32);         // (32, 32)
    out_gemm_wmma<<<grid, 256, 0, stream>>>(ctx, W_o, b_o, out, Mtiles);
}
